// Block_16501264351705
// MI455X (gfx1250) — compile-verified, matrix operand order checked by reading, not run on hardware
//
#include <hip/hip_runtime.h>
#include <math.h>

// ---------------- problem constants (match reference) ----------------
constexpr int Bb = 2, Tt = 4096, Cc = 512, Hh = 8, HSs = 64, Ff = 2048;
constexpr int Mm = Bb * Tt;                  // 8192 rows

// ---------------- WMMA types ----------------
typedef __attribute__((ext_vector_type(16))) __bf16        v16bf;
typedef __attribute__((ext_vector_type(8)))  float         v8f;
typedef __attribute__((ext_vector_type(4)))  unsigned int  v4u;
typedef __attribute__((ext_vector_type(4)))  float         v4f;

union Frag16 { v16bf bf; unsigned int u[8]; v4u q[2]; };
union Pack8  { v4u v; unsigned short s[8]; };

__device__ __forceinline__ v8f wmma_bf16(v16bf a, v16bf b, v8f c) {
  return __builtin_amdgcn_wmma_f32_16x16x32_bf16(false, a, false, b, (short)0, c,
                                                 false, false);
}

// bf16 fragment (ISA 7.12.2): dwords 0-3 hold K={0..7}+8*half, dwords 4-7 hold
// K={16..23}+8*half -> two contiguous 16B runs => two b128 loads per fragment.
__device__ __forceinline__ void frag_ld(Frag16& f, const unsigned short* base,
                                        int half) {
  const v4u* p = (const v4u*)(base + 8 * half);
  f.q[0] = p[0];        // u16 [8h .. 8h+7]
  f.q[1] = p[2];        // u16 [8h+16 .. 8h+23]
}

// raw LDS byte offset of a __shared__ object (AS3 pointers are 32-bit offsets)
__device__ __forceinline__ unsigned int lds_off(const void* p) {
  return (unsigned int)(unsigned long long)
      (__attribute__((address_space(3))) const void*)p;
}

// async Global -> LDS b128 copy (per-lane), tracked by ASYNCcnt
__device__ __forceinline__ void async_ld_b128(unsigned int ldsByteOff,
                                              const void* gaddr) {
  asm volatile("global_load_async_to_lds_b128 %0, %1, off"
               :: "v"(ldsByteOff), "v"(gaddr) : "memory");
}
__device__ __forceinline__ void wait_async0() {
  asm volatile("s_wait_asynccnt 0" ::: "memory");
}

__device__ __forceinline__ unsigned short f2bf(float f) {
  unsigned int u = __float_as_uint(f);
  u += 0x7FFFu + ((u >> 16) & 1u);           // round-to-nearest-even
  return (unsigned short)(u >> 16);
}

__device__ __forceinline__ float rmax16(float v) {
#pragma unroll
  for (int m = 1; m < 16; m <<= 1) v = fmaxf(v, __shfl_xor(v, m, 32));
  return v;
}
__device__ __forceinline__ float radd16(float v) {
#pragma unroll
  for (int m = 1; m < 16; m <<= 1) v += __shfl_xor(v, m, 32);
  return v;
}

// ---------------- weight conversion ----------------
__global__ void pack_headwise_bf16(const float* __restrict__ w,
                                   unsigned short* __restrict__ o) {
  const int total = Hh * Cc * HSs;
  for (int i = blockIdx.x * blockDim.x + threadIdx.x; i < total;
       i += gridDim.x * blockDim.x) {
    int d = i % HSs;
    int t = i / HSs;
    int c = t % Cc;
    int h = t / Cc;
    o[c * (Hh * HSs) + h * HSs + d] = f2bf(w[i]);
  }
}

__global__ void cast_bf16(const float* __restrict__ w,
                          unsigned short* __restrict__ o, int n) {
  for (int i = blockIdx.x * blockDim.x + threadIdx.x; i < n;
       i += gridDim.x * blockDim.x)
    o[i] = f2bf(w[i]);
}

// ---------------- LayerNorm (one wave32 per 512-wide row) ----------------
__global__ __launch_bounds__(256, 1) void layernorm_bf16(
    const float* __restrict__ x, const float* __restrict__ g,
    const float* __restrict__ b, unsigned short* __restrict__ o) {
  int lane = threadIdx.x & 31;
  int wid  = threadIdx.x >> 5;
  int row  = blockIdx.x * 8 + wid;
  const int c0 = lane * 16;                       // 16 contiguous elems per lane
  const v4f* xr = (const v4f*)(x + (size_t)row * Cc + c0);
  v4f vals[4];
  float s = 0.f;
#pragma unroll
  for (int i = 0; i < 4; ++i) {
    vals[i] = xr[i];
#pragma unroll
    for (int j = 0; j < 4; ++j) s += vals[i][j];
  }
#pragma unroll
  for (int m = 1; m < 32; m <<= 1) s += __shfl_xor(s, m, 32);
  float mu = s * (1.0f / Cc);
  float var = 0.f;
#pragma unroll
  for (int i = 0; i < 4; ++i)
#pragma unroll
    for (int j = 0; j < 4; ++j) { float d = vals[i][j] - mu; var += d * d; }
#pragma unroll
  for (int m = 1; m < 32; m <<= 1) var += __shfl_xor(var, m, 32);
  float inv = rsqrtf(var * (1.0f / Cc) + 1e-5f);
  const v4f* gr = (const v4f*)(g + c0);
  const v4f* br = (const v4f*)(b + c0);
  unsigned int* orow = (unsigned int*)(o + (size_t)row * Cc + c0);
#pragma unroll
  for (int i = 0; i < 4; ++i) {
    v4f gg = gr[i], bb = br[i];
#pragma unroll
    for (int j = 0; j < 2; ++j) {
      float e0 = (vals[i][2 * j]     - mu) * inv * gg[2 * j]     + bb[2 * j];
      float e1 = (vals[i][2 * j + 1] - mu) * inv * gg[2 * j + 1] + bb[2 * j + 1];
      orow[i * 2 + j] = (unsigned int)f2bf(e0) | ((unsigned int)f2bf(e1) << 16);
    }
  }
}

// ---------------- WMMA GEMM: C[M,N] = oscale*(A@B) [+bias][relu][+res] -------
// 256 threads = 8 waves (4 M x 2 N); block tile 128x128, wave tile 32x64.
// Double-buffered LDS (32KB). A-tile for step i+1 streams in via async
// global->LDS DMA (ASYNCcnt) while step i's WMMAs run; B-tile (needs a
// transpose) goes through registers. One s_barrier per K-step.
template <bool HAS_BIAS, bool RELU, bool HAS_RES, bool OUT_BF16>
__global__ __launch_bounds__(256, 1) void gemm_bf16(
    const unsigned short* __restrict__ A, const unsigned short* __restrict__ Bm,
    const float* __restrict__ bias, const float* __restrict__ res,
    float* __restrict__ outF, unsigned short* __restrict__ outB,
    int M, int N, int K, float oscale) {
  constexpr int BM = 128, BN = 128, BK = 32;
  __shared__ __attribute__((aligned(16))) unsigned short As[2][BM * BK];
  __shared__ __attribute__((aligned(16))) unsigned short Bs[2][BN * BK];
  const int tid  = threadIdx.x;
  const int lane = tid & 31, wid = tid >> 5;
  const int waveM = wid >> 1, waveN = wid & 1;
  const int bm = blockIdx.y * BM, bn = blockIdx.x * BN;
  const int ml = lane & 15, half = lane >> 4;

  // per-thread copy geometry (2 v4u each for A and B)
  int  idxA[2], kkB[2], n0B[2];
  const unsigned short* aSrc[2];
  const unsigned short* bSrc[2];
#pragma unroll
  for (int i = 0; i < 2; ++i) {
    int ia = i * 256 + tid;                   // 0..511
    idxA[i] = ia;
    aSrc[i] = A + (size_t)(bm + (ia >> 2)) * K + (ia & 3) * 8;
    kkB[i] = ia >> 4;
    n0B[i] = (ia & 15) * 8;
    bSrc[i] = Bm + (size_t)kkB[i] * N + bn + n0B[i];
  }
  const unsigned int aLds0 = lds_off(&As[0][0]);
  const unsigned int aLds1 = lds_off(&As[1][0]);

  v8f acc[2][4];
#pragma unroll
  for (int i = 0; i < 2; ++i)
#pragma unroll
    for (int j = 0; j < 4; ++j)
#pragma unroll
      for (int r = 0; r < 8; ++r) acc[i][j][r] = 0.f;

  v4u bReg[2];
  auto aAsync = [&](int buf, int k0) {        // A tile: DMA straight into LDS
    unsigned int base = buf ? aLds1 : aLds0;
#pragma unroll
    for (int i = 0; i < 2; ++i)
      async_ld_b128(base + idxA[i] * 16, aSrc[i] + k0);
  };
  auto bLoad = [&](int k0) {
#pragma unroll
    for (int i = 0; i < 2; ++i) bReg[i] = *(const v4u*)(bSrc[i] + (size_t)k0 * N);
    if (k0 + BK < K) {                        // speculative prefetch of step+2
      __builtin_prefetch(aSrc[0] + k0 + BK, 0, 1);
      __builtin_prefetch(bSrc[0] + (size_t)(k0 + BK) * N, 0, 1);
    }
  };
  auto bStore = [&](int buf) {                // transposed scatter [n][k]
#pragma unroll
    for (int i = 0; i < 2; ++i) {
      Pack8 pk; pk.v = bReg[i];
#pragma unroll
      for (int j = 0; j < 8; ++j) Bs[buf][(n0B[i] + j) * BK + kkB[i]] = pk.s[j];
    }
  };

  const int steps = K / BK;
  aAsync(0, 0);
  bLoad(0);
  bStore(0);
  wait_async0();
  __syncthreads();
  for (int it = 0; it < steps; ++it) {
    const int cur = it & 1;
    if (it + 1 < steps) {                     // stream next tiles during WMMAs
      aAsync(cur ^ 1, (it + 1) * BK);
      bLoad((it + 1) * BK);
    }

    Frag16 af[2];
#pragma unroll
    for (int mi = 0; mi < 2; ++mi)
      frag_ld(af[mi], As[cur] + (waveM * 32 + mi * 16 + ml) * BK, half);
#pragma unroll
    for (int ni = 0; ni < 4; ++ni) {
      Frag16 bfr;
      frag_ld(bfr, Bs[cur] + (waveN * 64 + ni * 16 + ml) * BK, half);
#pragma unroll
      for (int mi = 0; mi < 2; ++mi)
        acc[mi][ni] = wmma_bf16(af[mi].bf, bfr.bf, acc[mi][ni]);
    }

    if (it + 1 < steps) {
      bStore(cur ^ 1);                        // other buffer; safe post-barrier
      wait_async0();                          // A tile landed in LDS
    }
    __syncthreads();
  }

#pragma unroll
  for (int mi = 0; mi < 2; ++mi)
#pragma unroll
    for (int ni = 0; ni < 4; ++ni) {
      int n = bn + waveN * 64 + ni * 16 + ml;
      float bv = HAS_BIAS ? bias[n] : 0.f;
#pragma unroll
      for (int r = 0; r < 8; ++r) {
        int m = bm + waveM * 32 + mi * 16 + r + 8 * half;
        float v = acc[mi][ni][r] * oscale;
        if (HAS_BIAS) v += bv;
        if (RELU) v = fmaxf(v, 0.f);
        if (HAS_RES) v += res[(size_t)m * N + n];
        if (OUT_BF16) outB[(size_t)m * N + n] = f2bf(v);
        else          outF[(size_t)m * N + n] = v;
      }
    }
}

// ---------------- Flash attention ----------------
// Block = 128 query rows (8 waves x 16 rows), one (b,h). K tile streams in
// via async global->LDS DMA; V tile is transposed through registers.
// Q is pre-scaled by C^-0.5 in the projection GEMM.
__global__ __launch_bounds__(256, 1) void flash_attn(
    const unsigned short* __restrict__ qb, const unsigned short* __restrict__ kb,
    const unsigned short* __restrict__ vb, unsigned short* __restrict__ ob) {
  __shared__ __attribute__((aligned(16))) unsigned short Ks[32 * 64]; // [key][ch]
  __shared__ __attribute__((aligned(16))) unsigned short Vs[64 * 32]; // [ch][key]
  __shared__ __attribute__((aligned(16))) unsigned short Ps[8][16 * 32];
  const int tid = threadIdx.x;
  const int lane = tid & 31, wid = tid >> 5;
  const int ml = lane & 15, half = lane >> 4;
  const int bh = blockIdx.y;
  const int b = bh / Hh, h = bh % Hh;
  const int qtb = blockIdx.x;                 // 128-row block index
  const int qt = qtb * 8 + wid;               // this wave's 16-row tile
  const int q0 = qt * 16;
  const size_t baseRow = (size_t)b * Tt;
  const unsigned int ksLds = lds_off(Ks) + tid * 16;

  Frag16 qf[2];
  {
    const unsigned short* qrow = qb + (baseRow + q0 + ml) * Cc + h * HSs;
    frag_ld(qf[0], qrow, half);
    frag_ld(qf[1], qrow + 32, half);
  }

  float mrow[8], lrow[8];
  v8f O[4];
#pragma unroll
  for (int r = 0; r < 8; ++r) { mrow[r] = -3.0e38f; lrow[r] = 0.f; }
#pragma unroll
  for (int nt = 0; nt < 4; ++nt)
#pragma unroll
    for (int r = 0; r < 8; ++r) O[nt][r] = 0.f;

  const int npairs = 4 * qtb + 4;             // covers the block's causal range
  unsigned short* ps = Ps[wid];

  for (int kp = 0; kp < npairs; ++kp) {
    __syncthreads();                          // previous iter consumers done
    {
      int kk = tid >> 3, qi = tid & 7;
      // K tile [key][ch]: async DMA, one b128 per thread
      async_ld_b128(ksLds,
                    (const v4u*)(kb + (baseRow + kp * 32 + kk) * Cc + h * HSs) + qi);
      // V tile -> transposed [ch][key] through registers
      const v4u* vsrc =
          (const v4u*)(vb + (baseRow + kp * 32 + kk) * Cc + h * HSs);
      Pack8 pk; pk.v = vsrc[qi];
      int d0 = qi * 8;
#pragma unroll
      for (int j = 0; j < 8; ++j) Vs[(d0 + j) * 32 + kk] = pk.s[j];
    }
    wait_async0();
    __syncthreads();

    if (2 * kp <= qt) {                       // wave still inside causal range
      v8f s[2];
#pragma unroll
      for (int sub = 0; sub < 2; ++sub) {
        int kt = 2 * kp + sub;
        if (kt > qt) {
#pragma unroll
          for (int r = 0; r < 8; ++r) s[sub][r] = -3.0e38f;
          continue;
        }
        Frag16 kf[2];
        const unsigned short* krow = Ks + (sub * 16 + ml) * 64;
        frag_ld(kf[0], krow, half);
        frag_ld(kf[1], krow + 32, half);
        v8f sf;
#pragma unroll
        for (int r = 0; r < 8; ++r) sf[r] = 0.f;
        sf = wmma_bf16(qf[0].bf, kf[0].bf, sf);
        sf = wmma_bf16(qf[1].bf, kf[1].bf, sf);
#pragma unroll
        for (int r = 0; r < 8; ++r) {
          float v = sf[r];
          if (kt == qt && ml > r + 8 * half) v = -3.0e38f;  // diag causal mask
          s[sub][r] = v;
        }
      }

      // online softmax over the 32-key pair
      float alpha[8];
#pragma unroll
      for (int r = 0; r < 8; ++r) {
        float tm = rmax16(fmaxf(s[0][r], s[1][r]));
        float mn = fmaxf(mrow[r], tm);
        alpha[r] = __expf(mrow[r] - mn);
        float p0 = __expf(s[0][r] - mn);
        float p1 = __expf(s[1][r] - mn);
        s[0][r] = p0; s[1][r] = p1;
        lrow[r] = lrow[r] * alpha[r] + radd16(p0 + p1);
        mrow[r] = mn;
      }
#pragma unroll
      for (int nt = 0; nt < 4; ++nt)
#pragma unroll
        for (int r = 0; r < 8; ++r) O[nt][r] *= alpha[r];

      // stage P (C-fragment layout) into per-wave LDS, re-read as A-fragment
#pragma unroll
      for (int sub = 0; sub < 2; ++sub)
#pragma unroll
        for (int r = 0; r < 8; ++r)
          ps[(r + 8 * half) * 32 + 16 * sub + ml] = f2bf(s[sub][r]);
      asm volatile("s_wait_dscnt 0" ::: "memory");   // per-wave LDS RAW fence
      Frag16 pf;
      frag_ld(pf, ps + ml * 32, half);

      // P(16x32) @ V(32x64) -> O, four 16-col slabs, V frags from LDS (b128)
#pragma unroll
      for (int nt = 0; nt < 4; ++nt) {
        Frag16 vf;
        frag_ld(vf, Vs + (nt * 16 + ml) * 32, half);
        O[nt] = wmma_bf16(pf.bf, vf.bf, O[nt]);
      }
    }
  }

  // normalize and store (head-interleaved [BT, C] bf16)
#pragma unroll
  for (int r = 0; r < 8; ++r) {
    float inv = 1.0f / lrow[r];
    int m = q0 + r + 8 * half;
#pragma unroll
    for (int nt = 0; nt < 4; ++nt)
      ob[(baseRow + m) * Cc + h * HSs + nt * 16 + ml] = f2bf(O[nt][r] * inv);
  }
}

// ---------------- host launcher ----------------
extern "C" void kernel_launch(void* const* d_in, const int* in_sizes, int n_in,
                              void* d_out, int out_size, void* d_ws, size_t ws_size,
                              hipStream_t stream) {
  (void)in_sizes; (void)n_in; (void)out_size; (void)ws_size;
  const float* x   = (const float*)d_in[0];
  const float* wq  = (const float*)d_in[1];
  const float* wk  = (const float*)d_in[2];
  const float* wv  = (const float*)d_in[3];
  const float* wo  = (const float*)d_in[4];
  const float* bo  = (const float*)d_in[5];
  const float* w1  = (const float*)d_in[6];
  const float* b1  = (const float*)d_in[7];
  const float* w2  = (const float*)d_in[8];
  const float* b2  = (const float*)d_in[9];
  const float* g1  = (const float*)d_in[10];
  const float* be1 = (const float*)d_in[11];
  const float* g2  = (const float*)d_in[12];
  const float* be2 = (const float*)d_in[13];
  float* out = (float*)d_out;

  char* p = (char*)d_ws;
  auto alloc = [&](size_t bytes) -> void* {
    void* r = (void*)p;
    p += (bytes + 255) & ~(size_t)255;
    return r;
  };
  unsigned short* wq_bf = (unsigned short*)alloc((size_t)Cc * Cc * 2);
  unsigned short* wk_bf = (unsigned short*)alloc((size_t)Cc * Cc * 2);
  unsigned short* wv_bf = (unsigned short*)alloc((size_t)Cc * Cc * 2);
  unsigned short* wo_bf = (unsigned short*)alloc((size_t)Cc * Cc * 2);
  unsigned short* w1_bf = (unsigned short*)alloc((size_t)Cc * Ff * 2);
  unsigned short* w2_bf = (unsigned short*)alloc((size_t)Ff * Cc * 2);
  unsigned short* h1_bf = (unsigned short*)alloc((size_t)Mm * Cc * 2);
  unsigned short* q_bf  = (unsigned short*)alloc((size_t)Mm * Cc * 2);
  unsigned short* k_bf  = (unsigned short*)alloc((size_t)Mm * Cc * 2);
  unsigned short* v_bf  = (unsigned short*)alloc((size_t)Mm * Cc * 2);
  unsigned short* at_bf = (unsigned short*)alloc((size_t)Mm * Cc * 2);
  float*          x1    = (float*)alloc((size_t)Mm * Cc * 4);
  unsigned short* h2_bf = (unsigned short*)alloc((size_t)Mm * Cc * 2);
  unsigned short* ff_bf = (unsigned short*)alloc((size_t)Mm * Ff * 2);

  dim3 blk(256);

  pack_headwise_bf16<<<dim3(256), blk, 0, stream>>>(wq, wq_bf);
  pack_headwise_bf16<<<dim3(256), blk, 0, stream>>>(wk, wk_bf);
  pack_headwise_bf16<<<dim3(256), blk, 0, stream>>>(wv, wv_bf);
  cast_bf16<<<dim3(512),  blk, 0, stream>>>(wo, wo_bf, Cc * Cc);
  cast_bf16<<<dim3(2048), blk, 0, stream>>>(w1, w1_bf, Cc * Ff);
  cast_bf16<<<dim3(2048), blk, 0, stream>>>(w2, w2_bf, Ff * Cc);

  layernorm_bf16<<<dim3(Mm / 8), blk, 0, stream>>>(x, g1, be1, h1_bf);

  // QKV projections; Q pre-scaled by C^-0.5 (reference uses n_embed, not HS)
  dim3 gP(Cc / 128, Mm / 128);
  gemm_bf16<false, false, false, true><<<gP, blk, 0, stream>>>(
      h1_bf, wq_bf, nullptr, nullptr, nullptr, q_bf, Mm, Cc, Cc,
      rsqrtf((float)Cc));
  gemm_bf16<false, false, false, true><<<gP, blk, 0, stream>>>(
      h1_bf, wk_bf, nullptr, nullptr, nullptr, k_bf, Mm, Cc, Cc, 1.0f);
  gemm_bf16<false, false, false, true><<<gP, blk, 0, stream>>>(
      h1_bf, wv_bf, nullptr, nullptr, nullptr, v_bf, Mm, Cc, Cc, 1.0f);

  flash_attn<<<dim3(Tt / 128, Bb * Hh), blk, 0, stream>>>(q_bf, k_bf, v_bf,
                                                          at_bf);

  gemm_bf16<true, false, true, false><<<gP, blk, 0, stream>>>(
      at_bf, wo_bf, bo, x, x1, nullptr, Mm, Cc, Cc, 1.0f);

  layernorm_bf16<<<dim3(Mm / 8), blk, 0, stream>>>(x1, g2, be2, h2_bf);

  gemm_bf16<true, true, false, true><<<dim3(Ff / 128, Mm / 128), blk, 0, stream>>>(
      h2_bf, w1_bf, b1, nullptr, nullptr, ff_bf, Mm, Ff, Cc, 1.0f);

  gemm_bf16<true, false, true, false><<<gP, blk, 0, stream>>>(
      ff_bf, w2_bf, b2, x1, out, nullptr, Mm, Cc, Ff, 1.0f);
}
